// UniversalMultiANFIS_58669253263661
// MI455X (gfx1250) — matrix operand order repaired and checked by reading, not compiled, not run on hardware
//
#include <hip/hip_runtime.h>

typedef __attribute__((ext_vector_type(16))) _Float16 v16h;
typedef __attribute__((ext_vector_type(8)))  _Float16 v8h;
typedef __attribute__((ext_vector_type(2)))  _Float16 v2h;
typedef __attribute__((ext_vector_type(8)))  float    v8f;

#define NI    6
#define NM    3
#define NR    729          // 3^6 rules
#define NT    27           // K-tiles of 32; K-slot (t, c) -> rule t*27 + c (c<27)
#define SPB   16           // samples per wave (one 16-row WMMA tile)
#define WPB   4            // waves per block -> 64 samples per block
#define TPADH 40           // f16 table stride: 80B rows -> 16B-aligned, bank-balanced

// w[s][r] = T0[s][r/27] * T1[s][r%27];  K-slot mapping: rule = t*27 + (K%32)
// => per tile t: T0 factor == broadcast reg, T1 factor == loop-invariant regs.
// T1 pad entries [27..31] are zeroed, so dead K-slots produce 0 with no masking.
// GEMM per wave: D[16x16] = Wraw[16x864] @ Cons[864x16] (f16 in, f32 accum)
//   Cons cols 0..6 = cons_kp, cols 8..14 = cons_ki, col 15 = 1.0 -> row sums
__global__ __launch_bounds__(32 * WPB)
void anfis_wmma_kernel(const float* __restrict__ x,
                       const float* __restrict__ means,
                       const float* __restrict__ sigmas,
                       const float* __restrict__ cons_kp,
                       const float* __restrict__ cons_ki,
                       float* __restrict__ out,
                       int Bn)
{
    __shared__ __align__(32) _Float16 bFrag[NT * 32 * 16];   // 27 KB, all 4 waves
    __shared__ __align__(16) _Float16 T0h[WPB][SPB][TPADH];  // f16 mu0*mu1*mu2
    __shared__ __align__(16) _Float16 T1h[WPB][SPB][TPADH];  // f16 mu3*mu4*mu5
    __shared__ float muLDS[WPB][SPB * 18];
    __shared__ float xLDS [WPB][SPB * NI];
    __shared__ float dLDS [WPB][16 * 16];

    const int tid  = threadIdx.x;
    const int lane = tid & 31;
    const int wv   = tid >> 5;
    const int g0   = (blockIdx.x * WPB + wv) * SPB;

    __builtin_prefetch(cons_kp, 0, 3);
    __builtin_prefetch(cons_ki, 0, 3);

    // ---- Phase 0: load x tile (per wave: 16 samples x 6 inputs) ----
    for (int idx = lane; idx < SPB * NI; idx += 32) {
        int s  = idx / NI;
        int gs = g0 + s;
        xLDS[wv][idx] = (gs < Bn) ? x[gs * NI + (idx % NI)] : 0.0f;
    }
    __syncthreads();

    // ---- Phase 1a: Gaussian memberships (per wave, 288 values) ----
    for (int idx = lane; idx < SPB * 18; idx += 32) {
        int s   = idx / 18;
        int rem = idx % 18;
        int i   = rem / 3;
        int m   = rem % 3;
        float sg = sigmas[i * NM + m] + 1e-6f;
        float nd = (xLDS[wv][s * NI + i] - means[i * NM + m]) / sg;
        muLDS[wv][idx] = __expf(-0.5f * nd * nd);
    }

    // ---- Phase 1b: B fragments, packed f16x2 stores (block cooperates) ----
    // lane L: N = L%16, slot j -> c = (L<16?0:8)+(j&7)+((j&8)<<1), rule = t*27+c (c<27)
    for (int idx = tid; idx < NT * 256; idx += 32 * WPB) {
        int j  = (idx & 7) << 1;          // even slot of the pair
        int L  = (idx >> 3) & 31;
        int t  = idx >> 8;
        int n  = L & 15;
        int c0 = ((L >> 4) << 3) + (j & 7) + ((j & 8) << 1);
        v2h pk;
#pragma unroll
        for (int u = 0; u < 2; ++u) {
            int c = c0 + u;
            float v = 0.0f;
            if (c < 27) {
                int k = t * 27 + c;       // always < 729
                if (n < 7)                 v = cons_kp[k * 7 + n];
                else if (n >= 8 && n < 15) v = cons_ki[k * 7 + (n - 8)];
                else if (n == 15)          v = 1.0f;   // ones column -> sum(w)
            }
            pk[u] = (_Float16)v;
        }
        *(v2h*)&bFrag[(t * 32 + L) * 16 + j] = pk;
    }
    __syncthreads();   // mu ready

    // ---- Phase 2: partial-product tables -> f16 (per wave, 16*54 entries) ----
    for (int idx = lane; idx < SPB * 54; idx += 32) {
        int s  = idx / 54;
        int c  = idx % 54;
        int cc = (c < 27) ? c : c - 27;
        int base = (c < 27) ? 0 : 9;              // inputs 0-2 or 3-5
        const float* mu = &muLDS[wv][s * 18];
        float v = mu[base + cc / 9] * mu[base + 3 + (cc / 3) % 3] * mu[base + 6 + cc % 3];
        if (c < 27) T0h[wv][s][cc] = (_Float16)v;
        else        T1h[wv][s][cc] = (_Float16)v;
    }
    // zero T1 pad [27..31]: dead hi-half K-slots (c=27..31) then multiply to 0
    for (int idx = lane; idx < SPB * 5; idx += 32) {
        int s = idx / 5;
        T1h[wv][s][27 + idx % 5] = (_Float16)0.0f;
    }
    __syncthreads();   // tables + bFrag ready

    // ---- Phase 3: WMMA loop, A built in registers (8 pk_mul_f16 / tile) ----
    // lane: row m = lane&15, half hi -> slot j holds c = hi*8 + (j&7) + ((j&8)<<1)
    const int       m     = lane & 15;
    const bool      hiSel = (lane & 16) != 0;
    const _Float16* t0row = &T0h[wv][m][0];
    const _Float16* t1b   = &T1h[wv][m][0] + (hiSel ? 8 : 0);

    // preload all 27 T0 values (4 x ds_load_b128, 32 halves incl. pad)
    v8h w0q[4];
#pragma unroll
    for (int i = 0; i < 4; ++i) w0q[i] = *(const v8h*)(t0row + 8 * i);

    // loop-invariant T1 factors: c = base + {0..7} and base + {16..23}
    const v8h lo = *(const v8h*)(t1b);
    const v8h hi = *(const v8h*)(t1b + 16);

    v8f c = {};
#pragma unroll
    for (int t = 0; t < NT; ++t) {
        _Float16 w0 = w0q[t >> 3][t & 7];
        v8h ws  = {w0, w0, w0, w0, w0, w0, w0, w0};   // lowered to op_sel bcast

        v8h alo = lo * ws;     // 4 x v_pk_mul_f16
        v8h ahi = hi * ws;     // 4 x v_pk_mul_f16 (pad slots -> 0 via T1 zeros)
        v16h a  = __builtin_shufflevector(alo, ahi,
                      0, 1, 2, 3, 4, 5, 6, 7, 8, 9, 10, 11, 12, 13, 14, 15);

        v16h b = *(const v16h*)&bFrag[(t * 32 + lane) * 16];
        c = __builtin_amdgcn_wmma_f32_16x16x32_f16(
                /*neg_a=*/false, a, /*neg_b=*/false, b,
                /*c_mod=*/(short)0, c, /*reuse_a=*/false, /*reuse_b=*/false);
    }

    // ---- Phase 4: spill D (lane = col, VGPR v = row v + 8*(lane>=16)) ----
    {
        int col     = lane & 15;
        int rowBase = (lane >> 4) << 3;
#pragma unroll
        for (int v = 0; v < 8; ++v)
            dLDS[wv][(rowBase + v) * 16 + col] = c[v];
    }
    __syncthreads();

    // ---- Phase 5: normalize + x_bias dot + store (B,2) ----
    if (lane < 16) {
        float inv = 1.0f / (dLDS[wv][m * 16 + 15] + 1e-6f);
        float kp  = dLDS[wv][m * 16 + 6];    // bias column of cons_kp
        float ki  = dLDS[wv][m * 16 + 14];   // bias column of cons_ki
#pragma unroll
        for (int j = 0; j < 6; ++j) {
            float xv = xLDS[wv][m * NI + j];
            kp += xv * dLDS[wv][m * 16 + j];
            ki += xv * dLDS[wv][m * 16 + 8 + j];
        }
        int gs = g0 + m;
        if (gs < Bn) {
            out[gs * 2 + 0] = kp * inv;
            out[gs * 2 + 1] = ki * inv;
        }
    }
}

extern "C" void kernel_launch(void* const* d_in, const int* in_sizes, int n_in,
                              void* d_out, int out_size, void* d_ws, size_t ws_size,
                              hipStream_t stream) {
    const float* x       = (const float*)d_in[0];
    const float* means   = (const float*)d_in[1];
    const float* sigmas  = (const float*)d_in[2];
    const float* cons_kp = (const float*)d_in[3];
    const float* cons_ki = (const float*)d_in[4];
    // d_in[5] = rule_idx: digits implicit in the 27*a+b factorization

    int Bn   = in_sizes[0] / NI;
    int nblk = (Bn + SPB * WPB - 1) / (SPB * WPB);
    anfis_wmma_kernel<<<nblk, 32 * WPB, 0, stream>>>(x, means, sigmas, cons_kp, cons_ki,
                                                     (float*)d_out, Bn);
}
